// HandwritingSynthesis_27281632264537
// MI455X (gfx1250) — matrix operand order repaired and checked by reading, not compiled, not run on hardware
//
#include <hip/hip_runtime.h>
#include <cstdint>
#include <cstddef>

// ---------------------------------------------------------------------------
// Types for CDNA5 WMMA (wave32, 16x16x32 bf16 -> f32)
// ---------------------------------------------------------------------------
typedef __attribute__((ext_vector_type(16))) __bf16 v16bf;
typedef __attribute__((ext_vector_type(8)))  __bf16 v8bf;
typedef __attribute__((ext_vector_type(8)))  float  v8f;
typedef int v4i_vs __attribute__((vector_size(16)));   // matches async builtin param type

union ABu { v16bf v; v8bf h[2]; };

#define DEV __device__ __forceinline__

DEV float sigf(float x) { return 1.0f / (1.0f + __expf(-x)); }

DEV v8f wmma_bf16(const v16bf& a, const v16bf& b, const v8f& c) {
  return __builtin_amdgcn_wmma_f32_16x16x32_bf16(false, a, false, b, (short)0, c, false, false);
}

// ---------------------------------------------------------------------------
// CDNA5 async global<->LDS (ASYNCcnt) paths, guarded so compile stays green
// on toolchains without the builtins. Signature (probe-derived):
//   (AS1 v4i* global, AS3 v4i* lds, imm offset, imm cpol)
// ---------------------------------------------------------------------------
#if defined(__has_builtin)
#if __has_builtin(__builtin_amdgcn_global_load_async_to_lds_b128) && \
    __has_builtin(__builtin_amdgcn_s_wait_asynccnt)
#define HAVE_ASYNC_LOAD 1
#endif
#if __has_builtin(__builtin_amdgcn_global_store_async_from_lds_b128) && \
    __has_builtin(__builtin_amdgcn_s_wait_asynccnt)
#define HAVE_ASYNC_STORE 1
#endif
#endif

DEV void g2l_b128(void* lds, const void* g) {
#if defined(HAVE_ASYNC_LOAD)
  __builtin_amdgcn_global_load_async_to_lds_b128(
      (__attribute__((address_space(1))) v4i_vs*)g,
      (__attribute__((address_space(3))) v4i_vs*)lds, 0, 0);
#else
  *(uint4*)lds = *(const uint4*)g;
#endif
}

DEV void l2g_b128(void* g, const void* lds) {
#if defined(HAVE_ASYNC_STORE)
  __builtin_amdgcn_global_store_async_from_lds_b128(
      (__attribute__((address_space(1))) v4i_vs*)g,
      (__attribute__((address_space(3))) v4i_vs*)lds, 0, 0);
#else
  *(uint4*)g = *(const uint4*)lds;
#endif
}

DEV void wait_async() {
#if defined(HAVE_ASYNC_LOAD) || defined(HAVE_ASYNC_STORE)
  __builtin_amdgcn_s_wait_asynccnt(0);
#endif
}

// ---------------------------------------------------------------------------
// Problem constants
// ---------------------------------------------------------------------------
constexpr int BSZ  = 64;
constexpr int TLEN = 800;
constexpr int USEQ = 64;
constexpr int HDIM = 512;
constexpr int KMIX = 10;
constexpr int MMIX = 20;
constexpr int SENTC = 60;
constexpr int G4H  = 2048;             // 4*H gate width
constexpr int NT   = G4H / 16;         // 128 N-tiles of 16
constexpr int KT1  = (64 + HDIM) / 32; // 18  (padded inp(64) + h(512))
constexpr int KT2  = (576 + HDIM) / 32;// 34  ([h1|x|w|pad](576) + h2(512))
constexpr size_t NE = (size_t)BSZ * TLEN; // 51200 rows of output

// LSTM1 shared-memory layout (bytes)
constexpr int A1_STRIDE = 1168;                 // 576 bf16 cols + pad
constexpr int GSTR      = 2052;                 // gates row stride in floats
constexpr int OFF1_G    = 16 * A1_STRIDE;
constexpr int OFF1_C    = OFF1_G + 16 * GSTR * 4;
constexpr int OFF1_P    = OFF1_C + 16 * HDIM * 4;
constexpr int PBS       = 33;
constexpr int OFF1_AA   = OFF1_P + 16 * PBS * 4;
constexpr int OFF1_AB   = OFF1_AA + 16 * KMIX * 4;
constexpr int OFF1_KS   = OFF1_AB + 16 * KMIX * 4;
constexpr int PHIS      = 66;
constexpr int OFF1_PHI  = OFF1_KS + 16 * KMIX * 4;
constexpr int OFF1_W    = OFF1_PHI + 16 * PHIS * 4;
constexpr int SMEM1     = OFF1_W + 16 * SENTC * 4;          // 194880

// LSTM2 shared-memory layout (bytes)
constexpr int A2_STRIDE = 2192;                 // 1088 bf16 cols + pad
constexpr int OFF2_G    = 16 * A2_STRIDE;
constexpr int OFF2_C    = OFF2_G + 16 * GSTR * 4;
constexpr int SMEM2     = OFF2_C + 16 * HDIM * 4;           // 199168

// MDN shared memory
constexpr int PSTR  = 132;
constexpr int SMEM3 = 128 * PSTR * 4;                       // 67584

// Workspace layout (bytes)
constexpr size_t OFS_PW1  = 0;
constexpr size_t SZ_PW1   = (size_t)NT * KT1 * 1024;
constexpr size_t OFS_PW2  = OFS_PW1 + SZ_PW1;
constexpr size_t SZ_PW2   = (size_t)NT * KT2 * 1024;
constexpr size_t OFS_PWSW = OFS_PW2 + SZ_PW2;
constexpr size_t SZ_PWSW  = (size_t)2 * 16 * 1024;
constexpr size_t OFS_PWM1 = OFS_PWSW + SZ_PWSW;
constexpr size_t SZ_PWM   = (size_t)8 * 16 * 1024;
constexpr size_t OFS_PWM2 = OFS_PWM1 + SZ_PWM;
constexpr size_t OFS_H1   = OFS_PWM2 + SZ_PWM;
constexpr size_t SZ_HID   = NE * HDIM * 2;
constexpr size_t OFS_H2   = OFS_H1 + SZ_HID;
constexpr size_t OFS_WIN  = OFS_H2 + SZ_HID;
constexpr size_t SZ_WIN   = NE * 64 * 2;

// ---------------------------------------------------------------------------
// Weight packing: B-fragment layout for v_wmma_f32_16x16x32_bf16.
// Per 16(N)x32(K) tile: 32 lanes x 16 bf16 (32B) contiguous; lane l<16 holds
// col l, K = k0..k0+15; lane l>=16 holds col l-16, K = k0+16..k0+31.
// ---------------------------------------------------------------------------
DEV void pack_store(__bf16* out, int tile, int lane, const float* v) {
  v16bf f;
#pragma unroll
  for (int i = 0; i < 16; ++i) f[i] = (__bf16)v[i];
  *((v16bf*)(out) + ((size_t)tile * 32 + lane)) = f;
}

__global__ void pack_w1(const float* __restrict__ Wih1, const float* __restrict__ Whh1,
                        __bf16* __restrict__ out) {
  int tid = blockIdx.x * blockDim.x + threadIdx.x;
  if (tid >= NT * KT1 * 32) return;
  int lane = tid & 31, tile = tid >> 5;
  int nt = tile / KT1, kt = tile % KT1;
  int n = nt * 16 + (lane & 15);
  int k0 = kt * 32 + ((lane >> 4) << 4);
  float v[16];
#pragma unroll
  for (int i = 0; i < 16; ++i) {
    int k = k0 + i;
    v[i] = (k < 63) ? Wih1[(size_t)n * 63 + k]
         : (k < 64) ? 0.0f
                    : Whh1[(size_t)n * HDIM + (k - 64)];
  }
  pack_store(out, tile, lane, v);
}

__global__ void pack_w2(const float* __restrict__ Wih2, const float* __restrict__ Whh2,
                        __bf16* __restrict__ out) {
  int tid = blockIdx.x * blockDim.x + threadIdx.x;
  if (tid >= NT * KT2 * 32) return;
  int lane = tid & 31, tile = tid >> 5;
  int nt = tile / KT2, kt = tile % KT2;
  int n = nt * 16 + (lane & 15);
  int k0 = kt * 32 + ((lane >> 4) << 4);
  float v[16];
#pragma unroll
  for (int i = 0; i < 16; ++i) {
    int k = k0 + i;
    // A layout: [h1(512) | x(3) | w(60) | pad(1) | h2(512)]
    float x;
    if (k < 512)        x = Wih2[(size_t)n * 575 + (3 + k)];   // h1 block
    else if (k < 515)   x = Wih2[(size_t)n * 575 + (k - 512)]; // x block
    else if (k < 575)   x = Wih2[(size_t)n * 575 + k];         // w block
    else if (k < 576)   x = 0.0f;                              // pad
    else                x = Whh2[(size_t)n * HDIM + (k - 576)];
    v[i] = x;
  }
  pack_store(out, tile, lane, v);
}

__global__ void pack_wsw(const float* __restrict__ Wsw, __bf16* __restrict__ out) {
  int tid = blockIdx.x * blockDim.x + threadIdx.x;
  if (tid >= 2 * 16 * 32) return;
  int lane = tid & 31, tile = tid >> 5;
  int nt = tile / 16, kt = tile % 16;
  int n = nt * 16 + (lane & 15);
  int k0 = kt * 32 + ((lane >> 4) << 4);
  float v[16];
#pragma unroll
  for (int i = 0; i < 16; ++i)
    v[i] = (n < 30) ? Wsw[(size_t)n * HDIM + k0 + i] : 0.0f;
  pack_store(out, tile, lane, v);
}

__global__ void pack_wm(const float* __restrict__ Wmdn, __bf16* __restrict__ out1,
                        __bf16* __restrict__ out2) {
  int tid = blockIdx.x * blockDim.x + threadIdx.x;
  if (tid >= 8 * 16 * 32) return;
  int lane = tid & 31, tile = tid >> 5;
  int nt = tile / 16, kt = tile % 16;
  int n = nt * 16 + (lane & 15);
  int k0 = kt * 32 + ((lane >> 4) << 4);
  float v1[16], v2[16];
#pragma unroll
  for (int i = 0; i < 16; ++i) {
    int k = k0 + i;
    if (n < 121) {
      v1[i] = Wmdn[(size_t)n * 1536 + k];
      v2[i] = Wmdn[(size_t)n * 1536 + 512 + k] + Wmdn[(size_t)n * 1536 + 1024 + k];
    } else { v1[i] = 0.0f; v2[i] = 0.0f; }
  }
  pack_store(out1, tile, lane, v1);
  pack_store(out2, tile, lane, v2);
}

// ---------------------------------------------------------------------------
// A-fragment load helpers (16-bit A 16x32 layout):
// lane l<16: row l, K = k0..k0+15 (contiguous, matches B packing)
// lane l>=16: row l-16, K = k0+16..k0+31
// ---------------------------------------------------------------------------
DEV v16bf load_afrag_lds(const char* rowBase, int kbyte) {
  ABu a;
  a.h[0] = *(const v8bf*)(rowBase + kbyte);
  a.h[1] = *(const v8bf*)(rowBase + kbyte + 16);
  return a.v;
}

DEV v16bf load_afrag_g(const __bf16* p) {
  ABu a;
  a.h[0] = *(const v8bf*)(p);
  a.h[1] = *(const v8bf*)(p + 8);
  return a.v;
}

// ---------------------------------------------------------------------------
// Kernel 1: LSTM1 + attention window, persistent over T; 16 batch rows / block
// ---------------------------------------------------------------------------
__global__ __launch_bounds__(512) void lstm1_attn_kernel(
    const float* __restrict__ strks, const float* __restrict__ sents_m,
    const float* __restrict__ onehots, const float* __restrict__ w_prev0,
    const float* __restrict__ bih1, const float* __restrict__ bhh1,
    const float* __restrict__ bsw,
    const __bf16* __restrict__ pW1, const __bf16* __restrict__ pWsw,
    __bf16* __restrict__ hid1_bf, __bf16* __restrict__ win_bf) {
  extern __shared__ char smem[];
  char*  As    = smem;
  float* gates = (float*)(smem + OFF1_G);
  float* cbuf  = (float*)(smem + OFF1_C);
  float* pbuf  = (float*)(smem + OFF1_P);
  float* aA    = (float*)(smem + OFF1_AA);
  float* aB    = (float*)(smem + OFF1_AB);
  float* kapS  = (float*)(smem + OFF1_KS);
  float* phiS  = (float*)(smem + OFF1_PHI);
  float* wS    = (float*)(smem + OFF1_W);

  const int tid  = threadIdx.x;
  const int lane = tid & 31, wave = tid >> 5;
  const int l15  = lane & 15, hl = lane >> 4;
  const int b0   = blockIdx.x * 16;

  // ---- init persistent state ----
  for (int i = tid; i < 16 * A1_STRIDE / 4; i += 512) ((int*)As)[i] = 0;
  for (int i = tid; i < 16 * HDIM; i += 512) cbuf[i] = 0.0f;
  for (int i = tid; i < 16 * KMIX; i += 512) kapS[i] = 0.0f;
  for (int i = tid; i < 16 * SENTC; i += 512)
    wS[i] = w_prev0[(size_t)(b0 + (i / SENTC)) * SENTC + (i % SENTC)];
  __syncthreads();

  float bias[8];
#pragma unroll
  for (int i = 0; i < 8; ++i) {
    int n = (wave * 8 + i) * 16 + l15;
    bias[i] = bih1[n] + bhh1[n];
  }
  float bswl = 0.0f;
  if (wave < 2 && (wave * 16 + l15) < 30) bswl = bsw[wave * 16 + l15];

  const char* Arow = As + l15 * A1_STRIDE;

  for (int t = 0; t < TLEN; ++t) {
    // ---- stage inp = [x(3), w_prev(60), pad(1)] into A cols 0..63 ----
    for (int idx = tid; idx < 16 * 64; idx += 512) {
      int row = idx >> 6, c = idx & 63;
      float v;
      if (c < 3)       v = strks[((size_t)(b0 + row) * TLEN + t) * 3 + c];
      else if (c < 63) v = wS[row * SENTC + (c - 3)];
      else             v = 0.0f;
      *(__bf16*)(As + row * A1_STRIDE + c * 2) = (__bf16)v;
    }
    __syncthreads();

    // ---- gates GEMM: (16 x 576) * (576 x 2048), 8 N-tiles per wave ----
    v8f acc[8];
#pragma unroll
    for (int i = 0; i < 8; ++i) {
      v8f z;
#pragma unroll
      for (int j = 0; j < 8; ++j) z[j] = bias[i];
      acc[i] = z;
    }
    for (int kt = 0; kt < KT1; ++kt) {
      v16bf af = load_afrag_lds(Arow, (kt * 32 + hl * 16) * 2);
      if (kt + 1 < KT1)
        __builtin_prefetch(pW1 + ((size_t)((wave * 8) * KT1 + kt + 1) * 32 + lane) * 16, 0, 3);
#pragma unroll
      for (int i = 0; i < 8; ++i) {
        size_t tile = (size_t)(wave * 8 + i) * KT1 + kt;
        v16bf bf = *((const v16bf*)pW1 + tile * 32 + lane);
        acc[i] = wmma_bf16(af, bf, acc[i]);
      }
    }
#pragma unroll
    for (int i = 0; i < 8; ++i) {
      int n = (wave * 8 + i) * 16 + l15;
#pragma unroll
      for (int r = 0; r < 8; ++r) gates[(hl * 8 + r) * GSTR + n] = acc[i][r];
    }
    __syncthreads();

    // ---- LSTM cell update (16 x 512): h -> LDS (bf16, contiguous per row) --
    for (int idx = tid; idx < 16 * HDIM; idx += 512) {
      int row = idx >> 9, j = idx & 511;
      float gi = gates[row * GSTR + j];
      float gf = gates[row * GSTR + 512 + j];
      float gg = gates[row * GSTR + 1024 + j];
      float go = gates[row * GSTR + 1536 + j];
      float c = sigf(gf) * cbuf[row * HDIM + j] + sigf(gi) * tanhf(gg);
      float h = sigf(go) * tanhf(c);
      cbuf[row * HDIM + j] = c;
      *(__bf16*)(As + row * A1_STRIDE + (64 + j) * 2) = (__bf16)h;
    }
    __syncthreads();

    // ---- overlap: waves 0,1 run attention GEMM while waves 2..15 stream ----
    // ---- the h tile LDS -> global (async store-from-LDS, 128b chunks)   ----
    if (wave < 2) {
      v8f pa;
#pragma unroll
      for (int j = 0; j < 8; ++j) pa[j] = bswl;
      for (int kt = 0; kt < 16; ++kt) {
        v16bf af = load_afrag_lds(Arow, (64 + kt * 32 + hl * 16) * 2);
        v16bf bf = *((const v16bf*)pWsw + (size_t)(wave * 16 + kt) * 32 + lane);
        pa = wmma_bf16(af, bf, pa);
      }
      int n = wave * 16 + l15;
#pragma unroll
      for (int r = 0; r < 8; ++r) pbuf[(hl * 8 + r) * PBS + n] = pa[r];
    } else {
      for (int idx = tid - 64; idx < 16 * 64; idx += 448) {
        int row = idx >> 6, ch = idx & 63;
        void* src = As + row * A1_STRIDE + 128 + ch * 16;
        __bf16* dst = hid1_bf + ((size_t)(b0 + row) * TLEN + t) * HDIM + ch * 8;
        l2g_b128(dst, src);
      }
      wait_async();
    }
    __syncthreads();

    // ---- window parameters (alpha, beta, kappa) ----
    if (tid < 16 * KMIX) {
      int row = tid / KMIX, cc = tid % KMIX;
      float a  = __expf(pbuf[row * PBS + cc]);
      float bb = __expf(pbuf[row * PBS + 10 + cc]);
      float kp = kapS[row * KMIX + cc] + __expf(pbuf[row * PBS + 20 + cc]);
      kapS[row * KMIX + cc] = kp;
      aA[row * KMIX + cc] = a;
      aB[row * KMIX + cc] = bb;
    }
    __syncthreads();

    // ---- phi over U=64 (masked) ----
    for (int idx = tid; idx < 16 * USEQ; idx += 512) {
      int row = idx >> 6, u = idx & 63;
      float uu = (float)u, s = 0.0f;
#pragma unroll
      for (int cc = 0; cc < KMIX; ++cc) {
        float d = kapS[row * KMIX + cc] - uu;
        s += aA[row * KMIX + cc] * __expf(-aB[row * KMIX + cc] * d * d);
      }
      phiS[row * PHIS + u] = s * sents_m[(size_t)(b0 + row) * USEQ + u];
    }
    __syncthreads();

    // ---- w_t = phi @ onehots ----
    for (int idx = tid; idx < 16 * 64; idx += 512) {
      int row = idx >> 6, c = idx & 63;
      float w = 0.0f;
      if (c < SENTC) {
        const float* oh = onehots + ((size_t)(b0 + row) * USEQ) * SENTC + c;
        for (int u = 0; u < USEQ; ++u) w += phiS[row * PHIS + u] * oh[(size_t)u * SENTC];
        wS[row * SENTC + c] = w;
      }
      win_bf[((size_t)(b0 + row) * TLEN + t) * 64 + c] = (__bf16)w;
    }
    __syncthreads();
  }
}

// ---------------------------------------------------------------------------
// Kernel 2: LSTM2, persistent over T; A = [h1|x|w|pad | h2] (K = 1088)
// ---------------------------------------------------------------------------
__global__ __launch_bounds__(512) void lstm2_kernel(
    const float* __restrict__ strks, const float* __restrict__ bih2,
    const float* __restrict__ bhh2, const __bf16* __restrict__ pW2,
    const __bf16* __restrict__ hid1_bf, const __bf16* __restrict__ win_bf,
    __bf16* __restrict__ hid2_bf) {
  extern __shared__ char smem[];
  char*  As    = smem;
  float* gates = (float*)(smem + OFF2_G);
  float* cbuf  = (float*)(smem + OFF2_C);

  const int tid  = threadIdx.x;
  const int lane = tid & 31, wave = tid >> 5;
  const int l15  = lane & 15, hl = lane >> 4;
  const int b0   = blockIdx.x * 16;

  for (int i = tid; i < 16 * A2_STRIDE / 4; i += 512) ((int*)As)[i] = 0;
  for (int i = tid; i < 16 * HDIM; i += 512) cbuf[i] = 0.0f;
  __syncthreads();

  float bias[8];
#pragma unroll
  for (int i = 0; i < 8; ++i) {
    int n = (wave * 8 + i) * 16 + l15;
    bias[i] = bih2[n] + bhh2[n];
  }
  const char* Arow = As + l15 * A2_STRIDE;

  for (int t = 0; t < TLEN; ++t) {
    // ---- stage h1 (cols 0..511): async global -> LDS, 128b per chunk ----
    for (int idx = tid; idx < 16 * 64; idx += 512) {
      int row = idx >> 6, ch = idx & 63;
      const __bf16* src = hid1_bf + ((size_t)(b0 + row) * TLEN + t) * HDIM + ch * 8;
      g2l_b128(As + row * A2_STRIDE + ch * 16, src);
    }
    // ---- stage x (cols 512..514), w (515..574), pad (575) ----
    for (int idx = tid; idx < 16 * 64; idx += 512) {
      int row = idx >> 6, c = idx & 63;
      __bf16 v;
      if (c < 3)       v = (__bf16)strks[((size_t)(b0 + row) * TLEN + t) * 3 + c];
      else if (c < 63) v = win_bf[((size_t)(b0 + row) * TLEN + t) * 64 + (c - 3)];
      else             v = (__bf16)0.0f;
      *(__bf16*)(As + row * A2_STRIDE + (512 + c) * 2) = v;
    }
    wait_async();
    __syncthreads();

    v8f acc[8];
#pragma unroll
    for (int i = 0; i < 8; ++i) {
      v8f z;
#pragma unroll
      for (int j = 0; j < 8; ++j) z[j] = bias[i];
      acc[i] = z;
    }
    for (int kt = 0; kt < KT2; ++kt) {
      v16bf af = load_afrag_lds(Arow, (kt * 32 + hl * 16) * 2);
      if (kt + 1 < KT2)
        __builtin_prefetch(pW2 + ((size_t)((wave * 8) * KT2 + kt + 1) * 32 + lane) * 16, 0, 3);
#pragma unroll
      for (int i = 0; i < 8; ++i) {
        size_t tile = (size_t)(wave * 8 + i) * KT2 + kt;
        v16bf bf = *((const v16bf*)pW2 + tile * 32 + lane);
        acc[i] = wmma_bf16(af, bf, acc[i]);
      }
    }
#pragma unroll
    for (int i = 0; i < 8; ++i) {
      int n = (wave * 8 + i) * 16 + l15;
#pragma unroll
      for (int r = 0; r < 8; ++r) gates[(hl * 8 + r) * GSTR + n] = acc[i][r];
    }
    __syncthreads();

    for (int idx = tid; idx < 16 * HDIM; idx += 512) {
      int row = idx >> 9, j = idx & 511;
      float gi = gates[row * GSTR + j];
      float gf = gates[row * GSTR + 512 + j];
      float gg = gates[row * GSTR + 1024 + j];
      float go = gates[row * GSTR + 1536 + j];
      float c = sigf(gf) * cbuf[row * HDIM + j] + sigf(gi) * tanhf(gg);
      float h = sigf(go) * tanhf(c);
      cbuf[row * HDIM + j] = c;
      *(__bf16*)(As + row * A2_STRIDE + (576 + j) * 2) = (__bf16)h;
    }
    __syncthreads();

    // ---- stream h2 tile LDS -> global (async store-from-LDS) ----
    for (int idx = tid; idx < 16 * 64; idx += 512) {
      int row = idx >> 6, ch = idx & 63;
      void* src = As + row * A2_STRIDE + 1152 + ch * 16;
      __bf16* dst = hid2_bf + ((size_t)(b0 + row) * TLEN + t) * HDIM + ch * 8;
      l2g_b128(dst, src);
    }
    wait_async();
    __syncthreads();
  }
}

// ---------------------------------------------------------------------------
// Kernel 3: MDN head. params = h1@Wm1.T + h2@(Wm_b+Wm_c).T + bmdn, then
// sigmoid/softmax/exp/tanh transforms. 8 waves/block, 1 M-tile per wave.
// ---------------------------------------------------------------------------
__global__ __launch_bounds__(256) void mdn_kernel(
    const __bf16* __restrict__ hid1_bf, const __bf16* __restrict__ hid2_bf,
    const __bf16* __restrict__ pWm1, const __bf16* __restrict__ pWm2,
    const float* __restrict__ bmdn, float* __restrict__ out) {
  extern __shared__ char smem[];
  float* pL = (float*)smem;

  const int tid  = threadIdx.x;
  const int lane = tid & 31, wave = tid >> 5;
  const int l15  = lane & 15, hl = lane >> 4;
  const size_t mrow0 = ((size_t)blockIdx.x * 8 + wave) * 16;

  float bias[8];
#pragma unroll
  for (int i = 0; i < 8; ++i) {
    int n = i * 16 + l15;
    bias[i] = (n < 121) ? bmdn[n] : 0.0f;
  }
  v8f acc[8];
#pragma unroll
  for (int i = 0; i < 8; ++i) {
    v8f z;
#pragma unroll
    for (int j = 0; j < 8; ++j) z[j] = bias[i];
    acc[i] = z;
  }

  const __bf16* A1 = hid1_bf + mrow0 * HDIM;
  const __bf16* A2 = hid2_bf + mrow0 * HDIM;
  for (int kt = 0; kt < 32; ++kt) {
    const __bf16* Ab = (kt < 16) ? A1 : A2;
    int ktl = kt & 15;
    v16bf af = load_afrag_g(Ab + (size_t)l15 * HDIM + ktl * 32 + hl * 16);
    const __bf16* Wb = (kt < 16) ? pWm1 : pWm2;
#pragma unroll
    for (int i = 0; i < 8; ++i) {
      v16bf bf = *((const v16bf*)Wb + (size_t)(i * 16 + ktl) * 32 + lane);
      acc[i] = wmma_bf16(af, bf, acc[i]);
    }
  }

  float* pW = pL + wave * 16 * PSTR;
#pragma unroll
  for (int i = 0; i < 8; ++i)
#pragma unroll
    for (int r = 0; r < 8; ++r) pW[(hl * 8 + r) * PSTR + i * 16 + l15] = acc[i][r];
  __syncthreads();

  if (tid < 128) {
    const float* P = pL + tid * PSTR;
    size_t gm = (size_t)blockIdx.x * 128 + tid;
    out[gm] = sigf(P[120]);                    // eos
    float mx = P[0];
#pragma unroll
    for (int c = 1; c < MMIX; ++c) mx = fmaxf(mx, P[c]);
    float e[MMIX], s = 0.0f;
#pragma unroll
    for (int c = 0; c < MMIX; ++c) { e[c] = __expf(P[c] - mx); s += e[c]; }
    float inv = 1.0f / s;
    float* wo  = out + NE + gm * MMIX;
    float* m1o = out + NE + 1 * NE * MMIX + gm * MMIX;
    float* m2o = out + NE + 2 * NE * MMIX + gm * MMIX;
    float* s1o = out + NE + 3 * NE * MMIX + gm * MMIX;
    float* s2o = out + NE + 4 * NE * MMIX + gm * MMIX;
    float* ro  = out + NE + 5 * NE * MMIX + gm * MMIX;
#pragma unroll
    for (int c = 0; c < MMIX; ++c) {
      wo[c]  = e[c] * inv;
      m1o[c] = P[20 + c];
      m2o[c] = P[40 + c];
      s1o[c] = __expf(P[60 + c]);
      s2o[c] = __expf(P[80 + c]);
      ro[c]  = tanhf(P[100 + c]);
    }
  }
}

// ---------------------------------------------------------------------------
// Launch
// ---------------------------------------------------------------------------
extern "C" void kernel_launch(void* const* d_in, const int* in_sizes, int n_in,
                              void* d_out, int out_size, void* d_ws, size_t ws_size,
                              hipStream_t stream) {
  const float* strks   = (const float*)d_in[0];
  const float* sents_m = (const float*)d_in[3];
  const float* onehots = (const float*)d_in[4];
  const float* w_prev  = (const float*)d_in[5];
  const float* Wih1 = (const float*)d_in[6];
  const float* Whh1 = (const float*)d_in[7];
  const float* bih1 = (const float*)d_in[8];
  const float* bhh1 = (const float*)d_in[9];
  const float* Wih2 = (const float*)d_in[10];
  const float* Whh2 = (const float*)d_in[11];
  const float* bih2 = (const float*)d_in[12];
  const float* bhh2 = (const float*)d_in[13];
  const float* Wsw  = (const float*)d_in[14];
  const float* bsw  = (const float*)d_in[15];
  const float* Wmdn = (const float*)d_in[16];
  const float* bmdn = (const float*)d_in[17];
  float* out = (float*)d_out;

  char* ws = (char*)d_ws;
  __bf16* pW1  = (__bf16*)(ws + OFS_PW1);
  __bf16* pW2  = (__bf16*)(ws + OFS_PW2);
  __bf16* pWsw = (__bf16*)(ws + OFS_PWSW);
  __bf16* pWm1 = (__bf16*)(ws + OFS_PWM1);
  __bf16* pWm2 = (__bf16*)(ws + OFS_PWM2);
  __bf16* h1b  = (__bf16*)(ws + OFS_H1);
  __bf16* h2b  = (__bf16*)(ws + OFS_H2);
  __bf16* winb = (__bf16*)(ws + OFS_WIN);

  (void)hipFuncSetAttribute((const void*)lstm1_attn_kernel,
                            hipFuncAttributeMaxDynamicSharedMemorySize, SMEM1);
  (void)hipFuncSetAttribute((const void*)lstm2_kernel,
                            hipFuncAttributeMaxDynamicSharedMemorySize, SMEM2);
  (void)hipFuncSetAttribute((const void*)mdn_kernel,
                            hipFuncAttributeMaxDynamicSharedMemorySize, SMEM3);

  // weight packing (parallel, cheap)
  pack_w1<<<(NT * KT1 * 32 + 255) / 256, 256, 0, stream>>>(Wih1, Whh1, pW1);
  pack_w2<<<(NT * KT2 * 32 + 255) / 256, 256, 0, stream>>>(Wih2, Whh2, pW2);
  pack_wsw<<<(2 * 16 * 32 + 255) / 256, 256, 0, stream>>>(Wsw, pWsw);
  pack_wm<<<(8 * 16 * 32 + 255) / 256, 256, 0, stream>>>(Wmdn, pWm1, pWm2);

  // recurrent phases (batch split across 4 persistent workgroups)
  lstm1_attn_kernel<<<4, 512, SMEM1, stream>>>(strks, sents_m, onehots, w_prev,
                                               bih1, bhh1, bsw, pW1, pWsw, h1b, winb);
  lstm2_kernel<<<4, 512, SMEM2, stream>>>(strks, bih2, bhh2, pW2, h1b, winb, h2b);

  // parallel MDN head
  mdn_kernel<<<400, 256, SMEM3, stream>>>(h1b, h2b, pWm1, pWm2, bmdn, out);

  (void)in_sizes; (void)n_in; (void)out_size; (void)ws_size; (void)SZ_WIN;
}
